// NCF_74431783239698
// MI455X (gfx1250) — compile-verified, matrix operand order checked by reading, not run on hardware
//
#include <hip/hip_runtime.h>
#include <hip/hip_bf16.h>
#include <math.h>

// ---------------------------------------------------------------------------
// NCF + session-GAT recommender for MI455X (gfx1250, wave32, WMMA 16x16x32 f16)
// ---------------------------------------------------------------------------

typedef __attribute__((ext_vector_type(16))) _Float16 v16h;
typedef __attribute__((ext_vector_type(8)))  _Float16 v8h;
typedef __attribute__((ext_vector_type(8)))  float    v8f;

#define NUSER 100000
#define LSEQ  50
#define BTB   1024
#define BRB   4096

__device__ __forceinline__ v16h ld16h(const _Float16* p) {
  v8h lo = *(const v8h*)(p);
  v8h hi = *(const v8h*)(p + 16);
  v16h r;
#pragma unroll
  for (int i = 0; i < 8; ++i) { r[i] = lo[i]; r[8 + i] = hi[i]; }
  return r;
}

__device__ __forceinline__ v8f wmma32(v16h a, v16h b, v8f c) {
  // D = A(16x32 f16) * B(32x16 f16) + C(16x16 f32)
  return __builtin_amdgcn_wmma_f32_16x16x32_f16(false, a, false, b, (short)0, c,
                                                false, false);
}

// --------------------------- f32 -> f16 convert ----------------------------
__global__ void cvt_f16_kernel(const float* __restrict__ src,
                               _Float16* __restrict__ dst, int n) {
  int i = blockIdx.x * 256 + threadIdx.x;
  if (i < n) dst[i] = (_Float16)src[i];
}

// --------------------------- NCF gather ------------------------------------
__global__ void ncf_gather_kernel(const int* __restrict__ user,
                                  const int* __restrict__ item,
                                  const float* __restrict__ eu_gmf,
                                  const float* __restrict__ ei_gmf,
                                  const float* __restrict__ eu_mlp,
                                  const float* __restrict__ ei_mlp,
                                  _Float16* __restrict__ X0,
                                  float* __restrict__ gmf) {
  int b = blockIdx.x;
  int u = user[b];
  int it = item[b];
  for (int j = threadIdx.x; j < 512; j += 128) {
    float v = (j < 256) ? eu_mlp[(size_t)u * 256 + j]
                        : ei_mlp[(size_t)it * 256 + (j - 256)];
    X0[(size_t)b * 512 + j] = (_Float16)v;
  }
  if (threadIdx.x < 64) {
    int d = threadIdx.x;
    gmf[(size_t)b * 64 + d] =
        eu_gmf[(size_t)u * 64 + d] * ei_gmf[(size_t)it * 64 + d];
  }
}

// --------------------------- generic WMMA GEMM -----------------------------
// C(M x N) = relu(A(M x K) @ W(N x K)^T + bias), all tiles multiples of 16/32.
// One wave32 per 16x16 output tile.
__global__ void gemm16_f16_kernel(const _Float16* __restrict__ A, int lda,
                                  const _Float16* __restrict__ W,
                                  const float* __restrict__ bias,
                                  _Float16* __restrict__ C, int ldc,
                                  int K, int do_relu) {
  const int lane = threadIdx.x;
  const int m    = lane & 15;
  const int ksel = (lane >> 4) << 3;          // 0 or 8
  const int gm   = blockIdx.x * 16 + m;
  const int gn   = blockIdx.y * 16 + m;
  const _Float16* arow = A + (size_t)gm * lda;
  const _Float16* wrow = W + (size_t)gn * K;
  v8f c = {0.f, 0.f, 0.f, 0.f, 0.f, 0.f, 0.f, 0.f};
  for (int k0 = 0; k0 < K; k0 += 32) {
    v16h a = ld16h(arow + k0 + ksel);
    v16h b = ld16h(wrow + k0 + ksel);
    c = wmma32(a, b, c);
  }
  const int col = blockIdx.y * 16 + m;
  const float bv = bias[col];
#pragma unroll
  for (int r = 0; r < 8; ++r) {
    int row = blockIdx.x * 16 + r + ksel;
    float v = c[r] + bv;
    if (do_relu) v = v > 0.f ? v : 0.f;
    C[(size_t)row * ldc + col] = (_Float16)v;
  }
}

// --------------------------- rec head --------------------------------------
__global__ void rec_kernel(const float* __restrict__ gmf,
                           const _Float16* __restrict__ X3,
                           const float* __restrict__ pred_w,
                           const float* __restrict__ pred_b,
                           float* __restrict__ rec) {
  int i = blockIdx.x * 256 + threadIdx.x;
  if (i >= BRB) return;
  float acc = pred_b[0];
#pragma unroll 8
  for (int d = 0; d < 64; ++d) {
    acc += gmf[(size_t)i * 64 + d] * pred_w[d];
    acc += (float)X3[(size_t)i * 64 + d] * pred_w[64 + d];
  }
  rec[i] = acc;
}

// --------------------------- r2t: emb_cat -> tue ---------------------------
// Per 16-row tile: gather [GMF|MLP] 320-wide row, GEMM to 128 (ReLU),
// stage in LDS as f16, GEMM to 64 (ReLU). Writes tue f32 + f16 copy.
__global__ void r2t_kernel(const float* __restrict__ eu_gmf,
                           const float* __restrict__ eu_mlp,
                           const _Float16* __restrict__ w1h,  // 128 x 320
                           const float* __restrict__ b1,
                           const _Float16* __restrict__ w2h,  // 64 x 128
                           const float* __restrict__ b2,
                           float* __restrict__ tue,
                           _Float16* __restrict__ tueh) {
  __shared__ _Float16 s_h1[16][128];
  const int lane = threadIdx.x;
  const int m    = lane & 15;
  const int ksel = (lane >> 4) << 3;
  const int R    = blockIdx.x * 16;
  int row = R + m;
  if (row > NUSER) row = NUSER;

  v8f c[8];
#pragma unroll
  for (int nt = 0; nt < 8; ++nt)
    c[nt] = (v8f){0.f, 0.f, 0.f, 0.f, 0.f, 0.f, 0.f, 0.f};

  for (int k0 = 0; k0 < 320; k0 += 32) {
    v16h a;
#pragma unroll
    for (int h = 0; h < 2; ++h) {
      int kb = k0 + ksel + h * 16;
#pragma unroll
      for (int j = 0; j < 8; ++j) {
        int k = kb + j;
        float v = (k < 64) ? eu_gmf[(size_t)row * 64 + k]
                           : eu_mlp[(size_t)row * 256 + (k - 64)];
        a[h * 8 + j] = (_Float16)v;
      }
    }
#pragma unroll
    for (int nt = 0; nt < 8; ++nt) {
      v16h b = ld16h(w1h + (size_t)(nt * 16 + m) * 320 + k0 + ksel);
      c[nt] = wmma32(a, b, c[nt]);
    }
  }
#pragma unroll
  for (int nt = 0; nt < 8; ++nt) {
    int col = nt * 16 + m;
    float bv = b1[col];
#pragma unroll
    for (int r = 0; r < 8; ++r) {
      float v = c[nt][r] + bv;
      v = v > 0.f ? v : 0.f;
      s_h1[r + ksel][col] = (_Float16)v;
    }
  }
  __syncthreads();

  v8f c2[4];
#pragma unroll
  for (int nt = 0; nt < 4; ++nt)
    c2[nt] = (v8f){0.f, 0.f, 0.f, 0.f, 0.f, 0.f, 0.f, 0.f};
  for (int k0 = 0; k0 < 128; k0 += 32) {
    v16h a = ld16h(&s_h1[m][k0 + ksel]);
#pragma unroll
    for (int nt = 0; nt < 4; ++nt) {
      v16h b = ld16h(w2h + (size_t)(nt * 16 + m) * 128 + k0 + ksel);
      c2[nt] = wmma32(a, b, c2[nt]);
    }
  }
#pragma unroll
  for (int nt = 0; nt < 4; ++nt) {
    int col = nt * 16 + m;
    float bv = b2[col];
#pragma unroll
    for (int r = 0; r < 8; ++r) {
      int rr = R + r + ksel;
      if (rr <= NUSER) {
        float v = c2[nt][r] + bv;
        v = v > 0.f ? v : 0.f;
        tue[(size_t)rr * 64 + col]  = v;
        tueh[(size_t)rr * 64 + col] = (_Float16)v;
      }
    }
  }
}

// --------------------------- GAT + attention readout -----------------------
// One workgroup (256 threads) per session. All stages staged in LDS (~50KB).
__global__ void gat_kernel(const float* __restrict__ tue,
                           const int* __restrict__ inputs,
                           const int* __restrict__ mask,
                           const float* __restrict__ in_a,   // 4 x 128
                           const float* __restrict__ out_a,  // 128
                           const float* __restrict__ w,      // 256 x 64
                           const float* __restrict__ lin1_w, const float* __restrict__ lin1_b,
                           const float* __restrict__ lin2_w, const float* __restrict__ lin2_b,
                           const float* __restrict__ lin3_w,
                           const float* __restrict__ lt_w,   const float* __restrict__ lt_b,
                           const float* __restrict__ att_t,  // 128 x 2
                           _Float16* __restrict__ ah) {
  __shared__ float s_emb[LSEQ][64];
  __shared__ float s_mo[LSEQ][64];
  __shared__ float s_htmp[LSEQ][64];
  __shared__ float s_att[LSEQ][LSEQ];
  __shared__ float s_f1[LSEQ], s_f2[LSEQ];
  __shared__ float s_q1[64], s_alpha[LSEQ];
  __shared__ float s_pa[64], s_pavg[64], s_pf[64], s_att2[2];
  __shared__ int s_seql;

  const int b = blockIdx.x;
  const int tid = threadIdx.x;

  if (tid == 0) {
    int s = 0;
    for (int j = 0; j < LSEQ; ++j) s += mask[b * LSEQ + j];
    s_seql = s;
  }
  for (int t = tid; t < LSEQ * 64; t += 256) {
    int i = t >> 6, d = t & 63;
    s_emb[i][d] = tue[(size_t)inputs[b * LSEQ + i] * 64 + d];
    s_mo[i][d] = 0.f;
  }
  __syncthreads();
  const int seql = s_seql;

  // ---- 4 GAT heads, accumulated through the (256x64) projection w ----
  for (int k = 0; k < 4; ++k) {
    if (tid < LSEQ) {
      int i = tid;
      float acc = 0.f;
      for (int e = 0; e < 64; ++e) acc += s_emb[i][e] * in_a[k * 128 + e];
      s_f1[i] = acc;
    } else if (tid >= 64 && tid < 64 + LSEQ) {
      int i = tid - 64;
      float acc = 0.f;
      for (int e = 0; e < 64; ++e) acc += s_emb[i][e] * in_a[k * 128 + 64 + e];
      s_f2[i] = acc;
    }
    __syncthreads();
    if (tid < LSEQ) {  // masked row-softmax
      int i = tid;
      float f1i = s_f1[i];
      float mx = -3.0e38f;
      for (int j = 0; j < LSEQ; ++j) {
        float e;
        if (j < seql) { float t = f1i + s_f2[j]; e = t > 0.f ? t : 0.2f * t; }
        else e = -9.0e15f;
        s_att[i][j] = e;
        mx = e > mx ? e : mx;
      }
      float sum = 0.f;
      for (int j = 0; j < LSEQ; ++j) {
        float ex = expf(s_att[i][j] - mx);
        s_att[i][j] = ex;
        sum += ex;
      }
      float inv = 1.f / sum;
      for (int j = 0; j < LSEQ; ++j) s_att[i][j] *= inv;
    }
    __syncthreads();
    for (int t = tid; t < LSEQ * 64; t += 256) {  // h = elu(att @ emb)
      int i = t >> 6, d = t & 63;
      float acc = 0.f;
      for (int j = 0; j < LSEQ; ++j) acc += s_att[i][j] * s_emb[j][d];
      s_htmp[i][d] = acc > 0.f ? acc : (expf(acc) - 1.f);
    }
    __syncthreads();
    for (int t = tid; t < LSEQ * 64; t += 256) {  // mo += h_k @ w[k*64:...,:]
      int i = t >> 6, d = t & 63;
      float acc = 0.f;
      for (int e = 0; e < 64; ++e)
        acc += s_htmp[i][e] * w[(size_t)(k * 64 + e) * 64 + d];
      s_mo[i][d] += acc;
    }
    __syncthreads();
  }
  for (int t = tid; t < LSEQ * 64; t += 256) {  // mul_one = elu(mo)
    int i = t >> 6, d = t & 63;
    float v = s_mo[i][d];
    s_mo[i][d] = v > 0.f ? v : (expf(v) - 1.f);
  }
  __syncthreads();

  // ---- second GAT (out_a), no elu ----
  if (tid < LSEQ) {
    int i = tid;
    float acc = 0.f;
    for (int e = 0; e < 64; ++e) acc += s_mo[i][e] * out_a[e];
    s_f1[i] = acc;
  } else if (tid >= 64 && tid < 64 + LSEQ) {
    int i = tid - 64;
    float acc = 0.f;
    for (int e = 0; e < 64; ++e) acc += s_mo[i][e] * out_a[64 + e];
    s_f2[i] = acc;
  }
  __syncthreads();
  if (tid < LSEQ) {
    int i = tid;
    float f1i = s_f1[i];
    float mx = -3.0e38f;
    for (int j = 0; j < LSEQ; ++j) {
      float e;
      if (j < seql) { float t = f1i + s_f2[j]; e = t > 0.f ? t : 0.2f * t; }
      else e = -9.0e15f;
      s_att[i][j] = e;
      mx = e > mx ? e : mx;
    }
    float sum = 0.f;
    for (int j = 0; j < LSEQ; ++j) {
      float ex = expf(s_att[i][j] - mx);
      s_att[i][j] = ex;
      sum += ex;
    }
    float inv = 1.f / sum;
    for (int j = 0; j < LSEQ; ++j) s_att[i][j] *= inv;
  }
  __syncthreads();
  for (int t = tid; t < LSEQ * 64; t += 256) {  // hidden = att @ mo
    int i = t >> 6, d = t & 63;
    float acc = 0.f;
    for (int j = 0; j < LSEQ; ++j) acc += s_att[i][j] * s_mo[j][d];
    s_htmp[i][d] = acc;
  }
  __syncthreads();

  // ---- attention readout ----
  const int last = seql - 1;
  if (tid < 64) {
    int d = tid;
    float acc = lin1_b[d];
    for (int e = 0; e < 64; ++e) acc += s_htmp[last][e] * lin1_w[d * 64 + e];
    s_q1[d] = acc;
  }
  __syncthreads();
  if (tid < LSEQ) {
    int i = tid;
    float acc = 0.f;
    for (int d = 0; d < 64; ++d) {
      float q2 = lin2_b[d];
      for (int e = 0; e < 64; ++e) q2 += s_htmp[i][e] * lin2_w[d * 64 + e];
      float sg = 1.f / (1.f + expf(-(s_q1[d] + q2)));
      acc += sg * lin3_w[d];
    }
    s_alpha[i] = acc;
  }
  __syncthreads();
  if (tid < 64) {
    int d = tid;
    float acc = 0.f;
    for (int i = 0; i < LSEQ; ++i)
      acc += s_alpha[i] * s_htmp[i][d] * (float)mask[b * LSEQ + i];
    s_pa[d] = acc;
  } else if (tid < 128) {
    int d = tid - 64;
    float acc = 0.f;
    for (int i = 0; i < LSEQ; ++i)
      acc += (float)mask[b * LSEQ + i] * s_emb[i][d];
    s_pavg[d] = acc / (float)seql;
  }
  __syncthreads();
  if (tid < 64) {  // p_a_full = [p_a, ht] @ lt_w^T + lt_b
    int d = tid;
    float acc = lt_b[d];
    for (int e = 0; e < 64; ++e) acc += s_pa[e] * lt_w[d * 128 + e];
    for (int e = 0; e < 64; ++e) acc += s_htmp[last][e] * lt_w[d * 128 + 64 + e];
    s_pf[d] = acc;
  }
  __syncthreads();
  if (tid == 0) {
    float z0 = 0.f, z1 = 0.f;
    for (int e = 0; e < 64; ++e) {
      z0 += s_pf[e] * att_t[e * 2 + 0] + s_pavg[e] * att_t[(64 + e) * 2 + 0];
      z1 += s_pf[e] * att_t[e * 2 + 1] + s_pavg[e] * att_t[(64 + e) * 2 + 1];
    }
    float mz = z0 > z1 ? z0 : z1;
    float e0 = expf(z0 - mz), e1 = expf(z1 - mz);
    float inv = 1.f / (e0 + e1);
    s_att2[0] = e0 * inv;
    s_att2[1] = e1 * inv;
  }
  __syncthreads();
  if (tid < 64) {
    int d = tid;
    float v = s_pf[d] * s_att2[0] + s_pavg[d] * s_att2[1];
    ah[(size_t)b * 64 + d] = (_Float16)v;
  }
}

// --------------------------- scores = a @ b_emb^T --------------------------
// 8 waves per block, each wave owns one 16x16 tile (K=64 -> 2 WMMA).
__global__ void scores_kernel(const _Float16* __restrict__ ah,
                              const _Float16* __restrict__ tueh,
                              float* __restrict__ out) {
  const int wave = threadIdx.x >> 5;
  const int lane = threadIdx.x & 31;
  const int mt = blockIdx.x;
  const int nt = blockIdx.y * 8 + wave;
  if (nt >= 6250) return;
  const int m    = lane & 15;
  const int ksel = (lane >> 4) << 3;
  const _Float16* arow = ah + (size_t)(mt * 16 + m) * 64;
  const _Float16* brow = tueh + (size_t)(nt * 16 + m) * 64;
  v8f c = {0.f, 0.f, 0.f, 0.f, 0.f, 0.f, 0.f, 0.f};
#pragma unroll
  for (int k0 = 0; k0 < 64; k0 += 32) {
    v16h a = ld16h(arow + k0 + ksel);
    v16h b = ld16h(brow + k0 + ksel);
    c = wmma32(a, b, c);
  }
  const int col = nt * 16 + m;
#pragma unroll
  for (int r = 0; r < 8; ++r) {
    int row = mt * 16 + r + ksel;
    out[(size_t)row * NUSER + col] = c[r];
  }
}

// ---------------------------------------------------------------------------
extern "C" void kernel_launch(void* const* d_in, const int* in_sizes, int n_in,
                              void* d_out, int out_size, void* d_ws, size_t ws_size,
                              hipStream_t stream) {
  const int*   user   = (const int*)d_in[0];
  const int*   item   = (const int*)d_in[1];
  const int*   inputs = (const int*)d_in[2];
  const int*   mask   = (const int*)d_in[3];
  const float* eu_gmf = (const float*)d_in[4];
  const float* ei_gmf = (const float*)d_in[5];
  const float* eu_mlp = (const float*)d_in[6];
  const float* ei_mlp = (const float*)d_in[7];
  const float* w0     = (const float*)d_in[8];
  const float* b0     = (const float*)d_in[9];
  const float* w1     = (const float*)d_in[10];
  const float* b1     = (const float*)d_in[11];
  const float* w2     = (const float*)d_in[12];
  const float* b2     = (const float*)d_in[13];
  const float* pred_w = (const float*)d_in[14];
  const float* pred_b = (const float*)d_in[15];
  const float* r2t_w1 = (const float*)d_in[16];
  const float* r2t_b1 = (const float*)d_in[17];
  const float* r2t_w2 = (const float*)d_in[18];
  const float* r2t_b2 = (const float*)d_in[19];
  const float* in_a   = (const float*)d_in[20];
  const float* out_a  = (const float*)d_in[21];
  const float* w_gat  = (const float*)d_in[22];
  const float* lin1_w = (const float*)d_in[23];
  const float* lin1_b = (const float*)d_in[24];
  const float* lin2_w = (const float*)d_in[25];
  const float* lin2_b = (const float*)d_in[26];
  const float* lin3_w = (const float*)d_in[27];
  const float* lt_w   = (const float*)d_in[28];
  const float* lt_b   = (const float*)d_in[29];
  const float* att_t  = (const float*)d_in[30];

  char* ws = (char*)d_ws;
  size_t off = 0;
  auto alloc = [&](size_t bytes) -> void* {
    void* p = ws + off;
    off += (bytes + 255) & ~(size_t)255;
    return p;
  };
  _Float16* X0h  = (_Float16*)alloc((size_t)BRB * 512 * 2);
  _Float16* X1h  = (_Float16*)alloc((size_t)BRB * 256 * 2);
  _Float16* X2h  = (_Float16*)alloc((size_t)BRB * 128 * 2);
  _Float16* X3h  = (_Float16*)alloc((size_t)BRB * 64 * 2);
  float*    gmf  = (float*)   alloc((size_t)BRB * 64 * 4);
  _Float16* W0h  = (_Float16*)alloc((size_t)256 * 512 * 2);
  _Float16* W1h  = (_Float16*)alloc((size_t)128 * 256 * 2);
  _Float16* W2h  = (_Float16*)alloc((size_t)64 * 128 * 2);
  _Float16* R1h  = (_Float16*)alloc((size_t)128 * 320 * 2);
  _Float16* R2h  = (_Float16*)alloc((size_t)64 * 128 * 2);
  float*    tue  = (float*)   alloc((size_t)(NUSER + 1) * 64 * 4);
  _Float16* tueh = (_Float16*)alloc((size_t)(NUSER + 1) * 64 * 2);
  _Float16* ah   = (_Float16*)alloc((size_t)BTB * 64 * 2);

  float* rec    = (float*)d_out;
  float* scores = (float*)d_out + BRB;

  // Weight conversions (tiny, L2-resident afterwards).
  cvt_f16_kernel<<<(256 * 512 + 255) / 256, 256, 0, stream>>>(w0, W0h, 256 * 512);
  cvt_f16_kernel<<<(128 * 256 + 255) / 256, 256, 0, stream>>>(w1, W1h, 128 * 256);
  cvt_f16_kernel<<<(64 * 128 + 255) / 256, 256, 0, stream>>>(w2, W2h, 64 * 128);
  cvt_f16_kernel<<<(128 * 320 + 255) / 256, 256, 0, stream>>>(r2t_w1, R1h, 128 * 320);
  cvt_f16_kernel<<<(64 * 128 + 255) / 256, 256, 0, stream>>>(r2t_w2, R2h, 64 * 128);

  // NCF branch.
  ncf_gather_kernel<<<BRB, 128, 0, stream>>>(user, item, eu_gmf, ei_gmf,
                                             eu_mlp, ei_mlp, X0h, gmf);
  gemm16_f16_kernel<<<dim3(BRB / 16, 256 / 16), 32, 0, stream>>>(
      X0h, 512, W0h, b0, X1h, 256, 512, 1);
  gemm16_f16_kernel<<<dim3(BRB / 16, 128 / 16), 32, 0, stream>>>(
      X1h, 256, W1h, b1, X2h, 128, 256, 1);
  gemm16_f16_kernel<<<dim3(BRB / 16, 64 / 16), 32, 0, stream>>>(
      X2h, 128, W2h, b2, X3h, 64, 128, 1);
  rec_kernel<<<(BRB + 255) / 256, 256, 0, stream>>>(gmf, X3h, pred_w, pred_b, rec);

  // r2t transform over the full user table.
  r2t_kernel<<<(NUSER + 1 + 15) / 16, 32, 0, stream>>>(
      eu_gmf, eu_mlp, R1h, r2t_b1, R2h, r2t_b2, tue, tueh);

  // Session GAT + readout -> a (f16).
  gat_kernel<<<BTB, 256, 0, stream>>>(tue, inputs, mask, in_a, out_a, w_gat,
                                      lin1_w, lin1_b, lin2_w, lin2_b, lin3_w,
                                      lt_w, lt_b, att_t, ah);

  // scores = a @ tue[:-1]^T  (memory-bound: 410 MB of f32 stores).
  scores_kernel<<<dim3(BTB / 16, (6250 + 7) / 8), 256, 0, stream>>>(ah, tueh, scores);
}
